// OrderContrastiveLoss_11244224380936
// MI455X (gfx1250) — compile-verified, matrix operand order checked by reading, not run on hardware
//
#include <hip/hip_runtime.h>
#include <hip/hip_bf16.h>
#include <math.h>

// ---------------------------------------------------------------------------
// OrderContrastiveLoss on MI455X (gfx1250, wave32, WMMA)
//
// loss = sum_{a_i < a_j} softplus( (z_i - z_j) . W_{a_j} / (||z_i - z_j|| + eps) / T )
//        / (B*(B-1)),  W_a = unit(c_{a-1}-c_a) - unit(c_{a+1}-c_a)
//
// Dominant work: Gram matrix z @ z.T (4096x4096x512) via
// v_wmma_f32_16x16x32_bf16 with a 2-way bf16 split (hi/lo) for ~fp32 accuracy,
// fused with the pairwise softplus/mask/reduction (64MB Gram never touches HBM).
// 128x128 block tile, 8 waves x (32x64), double-buffered LDS staging using
// global_load_async_to_lds_b128 (ASYNCcnt) when available.
// ---------------------------------------------------------------------------

#define NUM_AGES 100
#define FEAT     512
#define BATCH    4096
#define EPSF     1e-6f
#define INV_T    10.0f

typedef __attribute__((ext_vector_type(16))) __bf16          v16bf;
typedef __attribute__((ext_vector_type(8)))  float           v8f;
typedef __attribute__((ext_vector_type(8)))  unsigned short  ushort8;
typedef __attribute__((ext_vector_type(4)))  unsigned int    uint4v;
typedef int v4i_vs __attribute__((vector_size(16)));   // matches builtin proto

#if defined(__HIP_DEVICE_COMPILE__) && defined(__gfx1250__) && \
    __has_builtin(__builtin_amdgcn_global_load_async_to_lds_b128)
#define USE_ASYNC_LDS 1
#endif

#if defined(__HIP_DEVICE_COMPILE__) && __has_builtin(__builtin_amdgcn_s_wait_asynccnt)
#define WAIT_ASYNC(n) __builtin_amdgcn_s_wait_asynccnt(n)
#elif defined(USE_ASYNC_LDS)
#define WAIT_ASYNC(n) asm volatile("s_wait_asynccnt %0" ::"n"(n) : "memory")
#else
#define WAIT_ASYNC(n)
#endif

// Load a 16-element bf16 fragment from an LDS tile stored [row][k], row stride
// 32 bf16. off0 is an element offset (16B aligned).
static __device__ __forceinline__ v16bf load_frag(const unsigned short* s, int off0) {
    v16bf f;
    ((ushort8*)&f)[0] = *(const ushort8*)(s + off0);
    ((ushort8*)&f)[1] = *(const ushort8*)(s + off0 + 16);
    return f;
}

// Stage four 128x32 bf16 tiles (Ah, Al, Bh, Bl) into LDS: 2048 16B chunks,
// 8 per thread (256 threads).
static __device__ __forceinline__ void stage_tiles(
    unsigned short (*dstTiles)[128 * 32],
    const __bf16* __restrict__ Zh, const __bf16* __restrict__ Zl,
    int i0, int j0, int kk, int tid) {
#pragma unroll
    for (int e = 0; e < 8; ++e) {
        const int q    = tid + (e << 8);
        const int tile = q >> 9;             // 0:Ah 1:Al 2:Bh 3:Bl
        const int rem  = q & 511;
        const int row  = rem >> 2;           // 0..127
        const int ch   = rem & 3;            // 16B chunk within the 32-k row
        const int base = (tile >= 2) ? j0 : i0;
        const __bf16* src = ((tile & 1) ? Zl : Zh)
                            + (size_t)(base + row) * FEAT + kk + ch * 8;
        unsigned short* dst = &dstTiles[tile][row * 32 + ch * 8];
#ifdef USE_ASYNC_LDS
        __builtin_amdgcn_global_load_async_to_lds_b128(
            (__attribute__((address_space(1))) v4i_vs*)(void*)src,
            (__attribute__((address_space(3))) v4i_vs*)(void*)dst,
            0, 0);
#else
        *(uint4v*)dst = *(const uint4v*)src;
#endif
    }
}

// ---------------------------------------------------------------------------
__global__ void zero_acc_kernel(double* acc) {
    if (threadIdx.x == 0) *acc = 0.0;
}

// ---------------------------------------------------------------------------
// Kernel 1: per-age direction difference W_a = vb_a - vf_a   [NUM_AGES][FEAT]
// ---------------------------------------------------------------------------
__global__ void prep_dirs_kernel(const float* __restrict__ proxies,
                                 float* __restrict__ W) {
    const int a = blockIdx.x;
    const int t = threadIdx.x;
    __shared__ float redf[128];
    __shared__ float redb[128];

    const int an = (a + 1 < NUM_AGES) ? a + 1 : NUM_AGES - 1;
    const int ap = (a - 1 >= 0) ? a - 1 : 0;
    const float* cur = proxies + (size_t)a  * FEAT;
    const float* nxt = proxies + (size_t)an * FEAT;
    const float* prv = proxies + (size_t)ap * FEAT;

    float df[4], db[4];
    float sf = 0.f, sb = 0.f;
#pragma unroll
    for (int e = 0; e < 4; ++e) {
        const int d = t + 128 * e;
        const float c = cur[d];
        const float f = nxt[d] - c;
        const float b = prv[d] - c;
        df[e] = f; db[e] = b;
        sf += f * f; sb += b * b;
    }
    redf[t] = sf; redb[t] = sb;
    __syncthreads();
    for (int s = 64; s > 0; s >>= 1) {
        if (t < s) { redf[t] += redf[t + s]; redb[t] += redb[t + s]; }
        __syncthreads();
    }
    const float inf_ = 1.f / (sqrtf(redf[0]) + EPSF);
    const float inb_ = 1.f / (sqrtf(redb[0]) + EPSF);
#pragma unroll
    for (int e = 0; e < 4; ++e) {
        const int d = t + 128 * e;
        W[(size_t)a * FEAT + d] = db[e] * inb_ - df[e] * inf_;
    }
}

// ---------------------------------------------------------------------------
// Kernel 2: per-row prep: sq = ||z_i||^2 (fp32), bf16 hi/lo split of z,
//           Zw[i][a] = z_i . W_a, dw[i] = Zw[i][age_i]
// ---------------------------------------------------------------------------
__global__ void prep_rows_kernel(const float* __restrict__ z,
                                 const int*   __restrict__ ages,
                                 const float* __restrict__ W,
                                 float* __restrict__ sq,
                                 float* __restrict__ Zw,
                                 float* __restrict__ dw,
                                 unsigned short* __restrict__ Zh_,
                                 unsigned short* __restrict__ Zl_) {
    const int i = blockIdx.x;
    const int t = threadIdx.x;
    __shared__ float zrow[FEAT];
    __shared__ float red[128];

    __bf16* Zh = (__bf16*)Zh_;
    __bf16* Zl = (__bf16*)Zl_;

    float s = 0.f;
#pragma unroll
    for (int e = 0; e < 4; ++e) {
        const int d = t + 128 * e;
        const float v = z[(size_t)i * FEAT + d];
        zrow[d] = v;
        s += v * v;
        const __bf16 h = (__bf16)v;
        Zh[(size_t)i * FEAT + d] = h;
        Zl[(size_t)i * FEAT + d] = (__bf16)(v - (float)h);
    }
    red[t] = s;
    __syncthreads();
    for (int s2 = 64; s2 > 0; s2 >>= 1) {
        if (t < s2) red[t] += red[t + s2];
        __syncthreads();
    }
    if (t == 0) sq[i] = red[0];

    if (t < NUM_AGES) {
        const float* Wr = W + (size_t)t * FEAT;
        float acc = 0.f;
#pragma unroll 4
        for (int d = 0; d < FEAT; ++d) acc += Wr[d] * zrow[d];
        Zw[(size_t)i * NUM_AGES + t] = acc;
        if (t == ages[i]) dw[i] = acc;
    }
}

// ---------------------------------------------------------------------------
// Kernel 3: fused Gram + pairwise loss.
// Grid 32x32 blocks; block tile 128x128; 256 threads = 8 waves arranged
// 4 (m) x 2 (n); each wave computes 32x64 -> acc[2][4] 16x16 tiles.
// Double-buffered LDS staging (2 x 32KB) with async loads when available.
// ---------------------------------------------------------------------------
__global__ void __launch_bounds__(256)
gram_loss_kernel(const float* __restrict__ sq,
                 const int*   __restrict__ ages,
                 const float* __restrict__ Zw,
                 const float* __restrict__ dw,
                 const unsigned short* __restrict__ Zh_,
                 const unsigned short* __restrict__ Zl_,
                 double* __restrict__ accum) {
    __shared__ __align__(16) unsigned short smem[2][4][128 * 32];

    const __bf16* Zh = (const __bf16*)Zh_;
    const __bf16* Zl = (const __bf16*)Zl_;

    const int tid  = threadIdx.x;
    const int wave = tid >> 5;
    const int lane = tid & 31;
    const int wm   = wave >> 1;          // 0..3 : 32-row strip
    const int wn   = wave & 1;           // 0..1 : 64-col strip
    const int i0 = blockIdx.y * 128;
    const int j0 = blockIdx.x * 128;

    v8f acc[2][4];
#pragma unroll
    for (int ms = 0; ms < 2; ++ms)
#pragma unroll
        for (int t = 0; t < 4; ++t)
#pragma unroll
            for (int r = 0; r < 8; ++r) acc[ms][t][r] = 0.f;

    // 16-bit A/B fragment offsets (ISA 7.12.2): lane<16 holds K {0..7,16..23},
    // lane>=16 holds K {8..15,24..31}; row index = lane%16 within the subtile.
    const int kb  = (lane >> 4) << 3;    // 0 or 8
    const int ln  = lane & 15;

    stage_tiles(smem[0], Zh, Zl, i0, j0, 0, tid);

    for (int s = 0; s < FEAT / 32; ++s) {
        if (s + 1 < FEAT / 32)
            stage_tiles(smem[(s + 1) & 1], Zh, Zl, i0, j0, 32 * (s + 1), tid);
#ifdef USE_ASYNC_LDS
        if (s + 1 < FEAT / 32) { WAIT_ASYNC(8); } else { WAIT_ASYNC(0); }
#endif
        __syncthreads();

        const unsigned short* tAh = smem[s & 1][0];
        const unsigned short* tAl = smem[s & 1][1];
        const unsigned short* tBh = smem[s & 1][2];
        const unsigned short* tBl = smem[s & 1][3];

        v16bf ah[2], al[2];
#pragma unroll
        for (int ms = 0; ms < 2; ++ms) {
            const int offA = (32 * wm + 16 * ms + ln) * 32 + kb;
            ah[ms] = load_frag(tAh, offA);
            al[ms] = load_frag(tAl, offA);
        }
#pragma unroll
        for (int t = 0; t < 4; ++t) {
            const int offB = (64 * wn + 16 * t + ln) * 32 + kb;
            const v16bf bh = load_frag(tBh, offB);
            const v16bf bl = load_frag(tBl, offB);
#pragma unroll
            for (int ms = 0; ms < 2; ++ms) {
                acc[ms][t] = __builtin_amdgcn_wmma_f32_16x16x32_bf16(
                    false, ah[ms], false, bh, (short)0, acc[ms][t], false, false);
                acc[ms][t] = __builtin_amdgcn_wmma_f32_16x16x32_bf16(
                    false, ah[ms], false, bl, (short)0, acc[ms][t], false, false);
                acc[ms][t] = __builtin_amdgcn_wmma_f32_16x16x32_bf16(
                    false, al[ms], false, bh, (short)0, acc[ms][t], false, false);
            }
        }
        __syncthreads();
    }

    // ---- fused epilogue: per-element pairwise loss ----
    // C/D layout: N = lane%16, M = vgpr + 8*(lane/16).
    const int N     = ln;
    const int Mhalf = (lane >> 4) << 3;

    int   ai8[2][8];
    float sqi8[2][8];
#pragma unroll
    for (int ms = 0; ms < 2; ++ms)
#pragma unroll
        for (int r = 0; r < 8; ++r) {
            const int i = i0 + 32 * wm + 16 * ms + Mhalf + r;
            ai8[ms][r]  = ages[i];
            sqi8[ms][r] = sq[i];
        }

    float lsum = 0.f;
#pragma unroll
    for (int t = 0; t < 4; ++t) {
        const int   j   = j0 + 64 * wn + 16 * t + N;
        const int   aj  = ages[j];
        const float sqj = sq[j];
        const float dwj = dw[j];
#pragma unroll
        for (int ms = 0; ms < 2; ++ms)
#pragma unroll
            for (int r = 0; r < 8; ++r) {
                const int   i  = i0 + 32 * wm + 16 * ms + Mhalf + r;
                const float G  = acc[ms][t][r];
                const float d2 = fmaxf(sqi8[ms][r] + sqj - 2.f * G, 0.f);
                const float dn = sqrtf(d2) + EPSF;
                const float nu = Zw[(size_t)i * NUM_AGES + aj] - dwj;
                const float x  = (nu / dn) * INV_T;
                const float sp = (x > 0.f) ? (x + log1pf(expf(-x)))
                                           : log1pf(expf(x));
                lsum += (ai8[ms][r] < aj) ? sp : 0.f;
            }
    }

    // wave32 reduction, one f64 atomic per wave
#pragma unroll
    for (int off = 16; off > 0; off >>= 1)
        lsum += __shfl_down(lsum, off, 32);
    if (lane == 0) atomicAdd(accum, (double)lsum);
}

// ---------------------------------------------------------------------------
__global__ void finalize_kernel(const double* acc, float* out) {
    if (threadIdx.x == 0)
        out[0] = (float)(*acc / ((double)BATCH * (double)(BATCH - 1)));
}

// ---------------------------------------------------------------------------
extern "C" void kernel_launch(void* const* d_in, const int* in_sizes, int n_in,
                              void* d_out, int out_size, void* d_ws, size_t ws_size,
                              hipStream_t stream) {
    const float* z       = (const float*)d_in[0];
    const int*   ages    = (const int*)d_in[1];
    const float* proxies = (const float*)d_in[2];

    char* ws = (char*)d_ws;
    size_t off = 0;
    double* accum = (double*)(ws + off); off += 256;
    float*  sq    = (float*)(ws + off);  off += (size_t)BATCH * 4;
    float*  dw    = (float*)(ws + off);  off += (size_t)BATCH * 4;
    float*  Zw    = (float*)(ws + off);  off += (size_t)BATCH * NUM_AGES * 4;
    float*  W     = (float*)(ws + off);  off += (size_t)NUM_AGES * FEAT * 4;
    unsigned short* Zh = (unsigned short*)(ws + off); off += (size_t)BATCH * FEAT * 2;
    unsigned short* Zl = (unsigned short*)(ws + off); off += (size_t)BATCH * FEAT * 2;

    hipLaunchKernelGGL(zero_acc_kernel, dim3(1), dim3(32), 0, stream, accum);
    hipLaunchKernelGGL(prep_dirs_kernel, dim3(NUM_AGES), dim3(128), 0, stream,
                       proxies, W);
    hipLaunchKernelGGL(prep_rows_kernel, dim3(BATCH), dim3(128), 0, stream,
                       z, ages, W, sq, Zw, dw, Zh, Zl);
    hipLaunchKernelGGL(gram_loss_kernel, dim3(32, 32), dim3(256), 0, stream,
                       sq, ages, Zw, dw, Zh, Zl, accum);
    hipLaunchKernelGGL(finalize_kernel, dim3(1), dim3(1), 0, stream,
                       accum, (float*)d_out);
}
